// GlobalPerformer_71674414235762
// MI455X (gfx1250) — matrix operand order repaired and checked by reading, not compiled, not run on hardware
//
#include <hip/hip_runtime.h>
#include <hip/hip_bf16.h>

#define B_  2
#define H_  8
#define L_  2048
#define D_  64
#define M_  64
#define BH_  (B_*H_)
#define BHL_ (BH_*L_)
#define SQ_SCALE 0.35355339059327373f   // sqrt(1/sqrt(64))
#define SM_SCALE 0.125f                 // 1/sqrt(64)
#define EPS_ 1e-4f
#define WIN_ 128

typedef __attribute__((ext_vector_type(16))) _Float16 v16h;
typedef __attribute__((ext_vector_type(8)))  _Float16 v8h;
typedef __attribute__((ext_vector_type(8)))  float    v8f;
typedef __attribute__((ext_vector_type(4)))  int      v4i;

#if __has_builtin(__builtin_amdgcn_global_load_async_to_lds_b128) && \
    __has_builtin(__builtin_amdgcn_s_wait_asynccnt)
#define USE_ASYNC 1
#else
#define USE_ASYNC 0
#endif

// 16-byte global->LDS copy; async DMA path on CDNA5 when available
static __device__ __forceinline__ void cp16(void* lds, const void* g) {
#if USE_ASYNC
  __builtin_amdgcn_global_load_async_to_lds_b128(
      (__attribute__((address_space(1))) v4i*)(g),
      (__attribute__((address_space(3))) v4i*)(lds), 0, 0);
#else
  *(uint4*)lds = *(const uint4*)g;
#endif
}
static __device__ __forceinline__ void cp16_wait() {
#if USE_ASYNC
  __builtin_amdgcn_s_wait_asynccnt(0);
#endif
}

static __device__ __forceinline__ v8f vzero8() {
  v8f z;
#pragma unroll
  for (int i = 0; i < 8; ++i) z[i] = 0.f;
  return z;
}

// A operand 16x32 f16 from row-major LDS (ISA 7.12.2): per lane the 16 halves
// are two contiguous 8-half runs: K in [8g, 8g+8) and [16+8g, 16+8g+8).
static __device__ __forceinline__ v16h ld_a(const _Float16* s, int ld, int row0, int col0) {
  int lane = threadIdx.x & 31;
  const _Float16* p = s + (row0 + (lane & 15)) * ld + col0 + ((lane >> 4) << 3);
  v8h lo = *(const v8h*)(p);
  v8h hi = *(const v8h*)(p + 16);
  return __builtin_shufflevector(lo, hi, 0, 1, 2, 3, 4, 5, 6, 7,
                                 8, 9, 10, 11, 12, 13, 14, 15);
}

// B operand 32x16 f16: lane%16 = N, halves = K in [8g, 8g+8) and [16+8g, 16+8g+8)
static __device__ __forceinline__ v16h ld_b(const _Float16* s, int ld, int row0, int col0) {
  int lane = threadIdx.x & 31;
  const _Float16* p = s + (row0 + ((lane >> 4) << 3)) * ld + col0 + (lane & 15);
  v16h r;
#pragma unroll
  for (int i = 0; i < 8; ++i) {
    r[i] = p[i * ld];
    r[8 + i] = p[(16 + i) * ld];
  }
  return r;
}

static __device__ __forceinline__ v8f wmma_f16(v16h a, v16h b, v8f c) {
  return __builtin_amdgcn_wmma_f32_16x16x32_f16(false, a, false, b, (short)0, c,
                                                false, false);
}

// ---------------------------------------------------------------------------
// Feature-map kernels (VALU; ~2 GFLOP total, bandwidth-trivial)
// ---------------------------------------------------------------------------

__global__ void __launch_bounds__(256)
k_logmax_kernel(const float* __restrict__ key, const float* __restrict__ proj,
                float* __restrict__ blockmax) {
  __shared__ float xsS[4][64];
  __shared__ float red[8];
  int tid = threadIdx.x;
  int rl = tid >> 6, m = tid & 63;
  long n = (long)blockIdx.x * 4 + rl;
  float x = key[n * 64 + m] * SQ_SCALE;
  xsS[rl][m] = x;
  __syncthreads();
  float xd = 0.f, dg = 0.f;
#pragma unroll 8
  for (int d = 0; d < 64; ++d) {
    float t = xsS[rl][d];
    xd += t * proj[d * 64 + m];
    dg += t * t;
  }
  float v = xd - 0.5f * dg;
  for (int off = 16; off > 0; off >>= 1) v = fmaxf(v, __shfl_down(v, off, 32));
  if ((tid & 31) == 0) red[tid >> 5] = v;
  __syncthreads();
  if (tid == 0) {
    float bm = red[0];
    for (int i = 1; i < 8; ++i) bm = fmaxf(bm, red[i]);
    blockmax[blockIdx.x] = bm;
  }
}

__global__ void __launch_bounds__(256)
k_ls_reduce(const float* __restrict__ blockmax, float* __restrict__ kls) {
  __shared__ float red[8];
  int tid = threadIdx.x;
  int base = blockIdx.x * 512;
  float v = -1e30f;
  for (int i = tid; i < 512; i += 256) v = fmaxf(v, blockmax[base + i]);
  for (int off = 16; off > 0; off >>= 1) v = fmaxf(v, __shfl_down(v, off, 32));
  if ((tid & 31) == 0) red[tid >> 5] = v;
  __syncthreads();
  if (tid == 0) {
    float m = red[0];
    for (int i = 1; i < 8; ++i) m = fmaxf(m, red[i]);
    kls[blockIdx.x] = m;
  }
}

// phi_k -> f16 workspace, both row-major [n][m] and transposed [m][n]
__global__ void __launch_bounds__(256)
kphi_kernel(const float* __restrict__ key, const float* __restrict__ proj,
            const float* __restrict__ kls, _Float16* __restrict__ phiK,
            _Float16* __restrict__ phiKT) {
  __shared__ float xsS[4][64];
  int tid = threadIdx.x;
  int rl = tid >> 6, m = tid & 63;
  long n = (long)blockIdx.x * 4 + rl;
  int bh = (int)(n >> 11);
  int l = (int)(n & (L_ - 1));
  float x = key[n * 64 + m] * SQ_SCALE;
  xsS[rl][m] = x;
  __syncthreads();
  float xd = 0.f, dg = 0.f;
#pragma unroll 8
  for (int d = 0; d < 64; ++d) {
    float t = xsS[rl][d];
    xd += t * proj[d * 64 + m];
    dg += t * t;
  }
  float lp = xd - 0.5f * dg;
  _Float16 ph = (_Float16)(__expf(lp - kls[bh]) + EPS_);
  phiK[n * 64 + m] = ph;
  phiKT[((long)bh * 64 + m) * L_ + l] = ph;   // L2-resident transpose scatter
}

// phi_q (per-row stabilizer) -> f16 workspace; also writes global_scale output
__global__ void __launch_bounds__(256)
qfeat_kernel(const float* __restrict__ query, const float* __restrict__ proj,
             const float* __restrict__ kls, _Float16* __restrict__ phiQ,
             float* __restrict__ out_scale) {
  __shared__ float xsS[4][64];
  __shared__ float red[4][2];
  int tid = threadIdx.x;
  int rl = tid >> 6, m = tid & 63;
  long n = (long)blockIdx.x * 4 + rl;
  int bh = (int)(n >> 11);
  float x = query[n * 64 + m] * SQ_SCALE;
  xsS[rl][m] = x;
  __syncthreads();
  float xd = 0.f, dg = 0.f;
#pragma unroll 8
  for (int d = 0; d < 64; ++d) {
    float t = xsS[rl][d];
    xd += t * proj[d * 64 + m];
    dg += t * t;
  }
  float lp = xd - 0.5f * dg;
  float v = lp;
  for (int off = 16; off > 0; off >>= 1) v = fmaxf(v, __shfl_down(v, off, 32));
  if ((m & 31) == 0) red[rl][m >> 5] = v;
  __syncthreads();
  float stab = fmaxf(red[rl][0], red[rl][1]);
  phiQ[n * 64 + m] = (_Float16)(__expf(lp - stab) + EPS_);
  if (m == 0) out_scale[n] = stab + kls[bh];
}

// global_out_cumsum: running feature sum + cross-lane dot, one block per (b,h)
__global__ void __launch_bounds__(64)
cumsum_kernel(const _Float16* __restrict__ phiQ, const _Float16* __restrict__ phiK,
              float* __restrict__ out_cs) {
  __shared__ float red2[2];
  int bh = blockIdx.x;
  int m = threadIdx.x;
  const _Float16* q = phiQ + (long)bh * L_ * 64;
  const _Float16* k = phiK + (long)bh * L_ * 64;
  float s = 0.f;
  for (int n = 0; n < L_; ++n) {
    s += (float)k[n * 64 + m];
    float p = (float)q[n * 64 + m] * s;
    for (int off = 16; off > 0; off >>= 1) p += __shfl_down(p, off, 32);
    if ((m & 31) == 0) red2[m >> 5] = p;
    __syncthreads();
    if (m == 0) out_cs[(long)bh * L_ + n] = red2[0] + red2[1];
    __syncthreads();
  }
}

// ---------------------------------------------------------------------------
// Causal linear attention scan: 1 WG per (b,h), 4 waves, chunk = 64.
// KV state in persistent WMMA accumulators; phiKT gives transpose-free B/A ops.
// ---------------------------------------------------------------------------
__global__ void __launch_bounds__(128)
linear_attn_kernel(const _Float16* __restrict__ phiQ, const _Float16* __restrict__ phiKT,
                   const float* __restrict__ value, float* __restrict__ out) {
  __shared__ __align__(16) _Float16 sQ[64 * 64];    // Q_c  [c][m]
  __shared__ __align__(16) _Float16 sKT[64 * 64];   // K_c^T [m][e]
  __shared__ __align__(16) _Float16 sV[64 * 64];    // V_c  [e][d]
  __shared__ __align__(16) _Float16 sKV[64 * 64];   // KV   [m][d]
  __shared__ __align__(16) _Float16 sAtt[64 * 64];  // attn [c][e]
  int bh = blockIdx.x;
  int tid = threadIdx.x, w = tid >> 5, lane = tid & 31;
  int g = lane >> 4, nn = lane & 15;
  const _Float16* qp = phiQ + (long)bh * L_ * 64;
  const _Float16* ktp = phiKT + (long)bh * 64 * L_;
  const float* vp = value + (long)bh * L_ * 64;
  float* op = out + (long)bh * L_ * 64;

  for (int i = tid; i < 64 * 64; i += 128) sKV[i] = (_Float16)0.f;
  v8f kvacc[4];
#pragma unroll
  for (int t = 0; t < 4; ++t) kvacc[t] = vzero8();
  __syncthreads();

  int r0 = w * 16;
  for (int c = 0; c < L_ / 64; ++c) {
    long base = (long)c * 64 * 64;
    // Q and K^T tiles via async-to-LDS DMA (16B per lane)
    for (int i = tid; i < 512; i += 128)
      cp16((uint4*)sQ + i, (const uint4*)(qp + base) + i);
    for (int i = tid; i < 512; i += 128) {
      int m = i >> 3, seg = i & 7;
      cp16((uint4*)sKT + i, (const uint4*)(ktp + (long)m * L_ + c * 64) + seg);
    }
    // V: f32 -> f16 conversion (VALU path)
    for (int i = tid; i < 1024; i += 128) {
      float4 vv = ((const float4*)(vp + base))[i];
      union { _Float16 h[4]; uint2 u; } pk;
      pk.h[0] = (_Float16)vv.x; pk.h[1] = (_Float16)vv.y;
      pk.h[2] = (_Float16)vv.z; pk.h[3] = (_Float16)vv.w;
      ((uint2*)sV)[i] = pk.u;
    }
    // prefetch next chunk while this one is consumed
    if (c + 1 < L_ / 64) {
      __builtin_prefetch(qp + base + 4096 + tid * 32, 0, 3);
      __builtin_prefetch(vp + base + 4096 + tid * 32, 0, 3);
    }
    cp16_wait();
    __syncthreads();

    // inter = Q_c @ KV
    v8f acc[4];
#pragma unroll
    for (int dt = 0; dt < 4; ++dt) acc[dt] = vzero8();
#pragma unroll
    for (int kc = 0; kc < 2; ++kc) {
      v16h a = ld_a(sQ, 64, r0, kc * 32);
#pragma unroll
      for (int dt = 0; dt < 4; ++dt)
        acc[dt] = wmma_f16(a, ld_b(sKV, 64, kc * 32, dt * 16), acc[dt]);
    }

    // attn = tril(Q_c @ K_c^T):  B[k=m][n=e] = sKT[m][e]
    v8f at[4];
#pragma unroll
    for (int et = 0; et < 4; ++et) at[et] = vzero8();
#pragma unroll
    for (int kc = 0; kc < 2; ++kc) {
      v16h a = ld_a(sQ, 64, r0, kc * 32);
#pragma unroll
      for (int et = 0; et < 4; ++et)
        at[et] = wmma_f16(a, ld_b(sKT, 64, kc * 32, et * 16), at[et]);
    }
#pragma unroll
    for (int et = 0; et < 4; ++et)
#pragma unroll
      for (int r = 0; r < 8; ++r) {
        int ci = r0 + r + 8 * g;
        int ce = et * 16 + nn;
        sAtt[ci * 64 + ce] = (_Float16)((ce <= ci) ? at[et][r] : 0.f);
      }
    __syncthreads();

    // intra = attn @ V_c, added into inter
#pragma unroll
    for (int kc = 0; kc < 2; ++kc) {
      v16h a = ld_a(sAtt, 64, r0, kc * 32);
#pragma unroll
      for (int dt = 0; dt < 4; ++dt)
        acc[dt] = wmma_f16(a, ld_b(sV, 64, kc * 32, dt * 16), acc[dt]);
    }
#pragma unroll
    for (int dt = 0; dt < 4; ++dt)
#pragma unroll
      for (int r = 0; r < 8; ++r) {
        int row = c * 64 + r0 + r + 8 * g;
        op[(long)row * 64 + dt * 16 + nn] = acc[dt][r];
      }

    // KV += K_c^T @ V_c : A[i=m][k=e] = sKT[m][e] (row-major -> vector ld_a)
#pragma unroll
    for (int ec = 0; ec < 2; ++ec) {
      v16h a = ld_a(sKT, 64, w * 16, ec * 32);
#pragma unroll
      for (int dt = 0; dt < 4; ++dt)
        kvacc[dt] = wmma_f16(a, ld_b(sV, 64, ec * 32, dt * 16), kvacc[dt]);
    }
    __syncthreads();
#pragma unroll
    for (int dt = 0; dt < 4; ++dt)
#pragma unroll
      for (int r = 0; r < 8; ++r) {
        int mrow = w * 16 + r + 8 * g;
        sKV[mrow * 64 + dt * 16 + nn] = (_Float16)kvacc[dt][r];
      }
    __syncthreads();
  }
}

// ---------------------------------------------------------------------------
// global_qk: banded WMMA scores + float4 zero fill of the off-band region.
// 1 WG per (b,h,128 rows); disjoint tile ownership -> no write races.
// ---------------------------------------------------------------------------
__global__ void __launch_bounds__(256)
local_attn_kernel(const _Float16* __restrict__ phiQ, const _Float16* __restrict__ phiKT,
                  float* __restrict__ outqk) {
  __shared__ __align__(16) _Float16 sQ[128 * 64];    // rows [R0, R0+128)
  __shared__ __align__(16) _Float16 sKT[64 * 256];   // [m][e], e in [R0-128, R0+128)
  int rb = blockIdx.x;
  int bh = blockIdx.y;
  int R0 = rb * 128;
  int tid = threadIdx.x, w = tid >> 5, lane = tid & 31;
  int g = lane >> 4, nn = lane & 15;
  const _Float16* qp = phiQ + (long)bh * L_ * 64;
  const _Float16* ktp = phiKT + (long)bh * 64 * L_;
  float* op = outqk + (long)bh * L_ * L_;

  for (int i = tid; i < 1024; i += 256)
    cp16((uint4*)sQ + i, (const uint4*)(qp + (long)R0 * 64) + i);
  for (int i = tid; i < 2048; i += 256) {
    int m = i >> 5, seg = i & 31;
    int e0 = R0 - 128 + seg * 8;
    if (e0 >= 0) cp16((uint4*)sKT + i, (const uint4*)(ktp + (long)m * L_ + e0));
  }
  cp16_wait();
  __syncthreads();

  int r0 = w * 16;
  int grow0 = R0 + r0;
  int jt_lo = (grow0 - WIN_) >> 4;       // exact: multiple of 16
  if (jt_lo < 0) jt_lo = 0;
  int jt_hi = grow0 >> 4;

  // zero fill this wave's 16 rows, skipping its band tiles
  float4 z = make_float4(0.f, 0.f, 0.f, 0.f);
  for (int r = 0; r < 16; ++r) {
    long rowoff = (long)(grow0 + r) * L_;
#pragma unroll
    for (int it = 0; it < 16; ++it) {
      int col = (lane + it * 32) * 4;
      int ct = col >> 4;
      if (ct >= jt_lo && ct <= jt_hi) continue;
      *(float4*)(op + rowoff + col) = z;
    }
  }

  // band tiles: scores = (phi_q @ phi_k^T) * SM_SCALE, band mask + relu
  for (int jt = jt_lo; jt <= jt_hi; ++jt) {
    int jcol = jt * 16 - R0 + 128;       // e-offset inside sKT
    v8f accv = vzero8();
#pragma unroll
    for (int kc = 0; kc < 2; ++kc) {
      v16h a = ld_a(sQ, 64, r0, kc * 32);
      v16h b = ld_b(sKT, 256, kc * 32, jcol);   // B[k=m][n=e]
      accv = wmma_f16(a, b, accv);
    }
#pragma unroll
    for (int r = 0; r < 8; ++r) {
      int i = grow0 + r + 8 * g;
      int j = jt * 16 + nn;
      float s = accv[r] * SM_SCALE;
      op[(long)i * L_ + j] = (j <= i && j >= i - WIN_ && s > 0.f) ? s : 0.f;
    }
  }
}

// ---------------------------------------------------------------------------
extern "C" void kernel_launch(void* const* d_in, const int* in_sizes, int n_in,
                              void* d_out, int out_size, void* d_ws, size_t ws_size,
                              hipStream_t stream) {
  (void)in_sizes; (void)n_in; (void)out_size; (void)ws_size;
  const float* query = (const float*)d_in[0];
  const float* key_  = (const float*)d_in[1];
  const float* value = (const float*)d_in[2];
  const float* proj  = (const float*)d_in[3];

  float* out = (float*)d_out;
  float* out_gout  = out;                         // (B,H,L,D)
  float* out_cs    = out + (long)BHL_ * D_;       // (B,H,L)
  float* out_scale = out_cs + BHL_;               // (B,H,L,1)
  float* out_qk    = out_scale + BHL_;            // (B,H,L,L)

  _Float16* phiQ  = (_Float16*)d_ws;                    // 4 MB
  _Float16* phiK  = phiQ + (long)BHL_ * M_;             // 4 MB
  _Float16* phiKT = phiK + (long)BHL_ * M_;             // 4 MB, [bh][m][l]
  float* blockmax = (float*)(phiKT + (long)BHL_ * M_);  // 32 KB
  float* kls = blockmax + (BHL_ / 4);                   // 16 floats

  int nfb = BHL_ / 4;   // 8192 feature blocks (4 rows each)
  k_logmax_kernel<<<nfb, 256, 0, stream>>>(key_, proj, blockmax);
  k_ls_reduce<<<BH_, 256, 0, stream>>>(blockmax, kls);
  kphi_kernel<<<nfb, 256, 0, stream>>>(key_, proj, kls, phiK, phiKT);
  qfeat_kernel<<<nfb, 256, 0, stream>>>(query, proj, kls, phiQ, out_scale);
  cumsum_kernel<<<BH_, 64, 0, stream>>>(phiQ, phiK, out_cs);
  linear_attn_kernel<<<BH_, 128, 0, stream>>>(phiQ, phiKT, value, out_gout);
  local_attn_kernel<<<dim3(L_ / 128, BH_), 256, 0, stream>>>(phiQ, phiKT, out_qk);
}
